// SwinTransformerBlock3D_13022340842203
// MI455X (gfx1250) — compile-verified, matrix operand order checked by reading, not run on hardware
//
#include <hip/hip_runtime.h>
#include <hip/hip_bf16.h>
#include <math.h>

// ---------------------------------------------------------------------------
// Swin Transformer 3D block for MI455X (gfx1250), wave32 + WMMA f16->f32.
// All GEMMs use v_wmma_f32_16x16x32_f16. Layouts are chosen so every WMMA
// fragment load is two contiguous 16-byte loads per lane:
//   A frag (16x32, row-major src, ld halfs): lane L -> row L%16,
//     halfs [0..7] = K {base..base+7} for lanes 0-15 / {base+8..15} lanes 16-31,
//     halfs [8..15] = +16.
//   B frag (32x16): identical addressing against a [N][K] (transposed) matrix.
// C/D frag: lane L holds column L%16, rows (L<16 ? 0..7 : 8..15) in v8f.
// GEMM kernels: each wave owns a 32x16 output tile (two M-tiles sharing one
// B fragment per K-step) -> 1.5 fragment loads per WMMA instead of 2.
// ---------------------------------------------------------------------------

typedef _Float16 v16h __attribute__((ext_vector_type(16)));
typedef _Float16 v8h  __attribute__((ext_vector_type(8)));
typedef _Float16 v4h  __attribute__((ext_vector_type(4)));
typedef float    v8f  __attribute__((ext_vector_type(8)));

#define C_DIM  128
#define NHEADS 4
#define HD     32
#define NTOK   98      // 2*7*7 tokens per window
#define NPAD   112     // padded to 7*16 for WMMA tiling
#define NWIN   1024    // 4 batches * (8/2)*(56/7)*(56/7)
#define TT     100352  // 4*8*56*56 tokens
#define MLPH   512

__device__ __forceinline__ v8f wmma16(v16h a, v16h b, v8f c) {
  return __builtin_amdgcn_wmma_f32_16x16x32_f16(false, a, false, b, (short)0, c,
                                                false, false);
}

// Load a 16x32 fragment (A-style) / 32x16 fragment (B-style vs transposed src).
__device__ __forceinline__ v16h ld_frag(const _Float16* p, int ld, int lane) {
  const _Float16* q = p + (size_t)(lane & 15) * ld + ((lane >> 4) << 3);
  v8h lo = *(const v8h*)q;
  v8h hi = *(const v8h*)(q + 16);
  v16h r;
#pragma unroll
  for (int i = 0; i < 8; ++i) { r[i] = lo[i]; r[8 + i] = hi[i]; }
  return r;
}

// ---------------- kernel 0: f32 weight -> transposed f16 [N][K] -------------
__global__ void k_transpose_w(const float* __restrict__ src,
                              _Float16* __restrict__ dst, int K, int Ncol) {
  int id = blockIdx.x * 256 + threadIdx.x;
  if (id >= K * Ncol) return;
  int n = id / K, kk = id - n * K;
  dst[id] = (_Float16)src[kk * Ncol + n];   // dst[n][k] = src[k][n]
}

// ------------- kernel 1: LN1 + roll(-1,-3,-3) + window partition ------------
// One wave per padded window row; lanes cover 4 channels each (128 = 32*4).
__global__ void k_ln1_window(const float* __restrict__ x,
                             const float* __restrict__ g,
                             const float* __restrict__ bt,
                             _Float16* __restrict__ xw) {
  int gid  = blockIdx.x * 8 + (threadIdx.x >> 5);
  int lane = threadIdx.x & 31;
  int win = gid / NPAD, p = gid - win * NPAD;
  _Float16* dst = xw + (size_t)gid * C_DIM + lane * 4;
  if (p >= NTOK) { v4h z = {}; *(v4h*)dst = z; return; }
  int b = win >> 8, rem = win & 255;
  int dblk = rem >> 6, hblk = (rem >> 3) & 7, wblk = rem & 7;
  int id = p / 49, r2 = p - id * 49, ih = r2 / 7, iw = r2 - ih * 7;
  int ds = (dblk * 2 + id + 1) & 7;                 // roll by -SHIFT
  int hs = hblk * 7 + ih + 3; if (hs >= 56) hs -= 56;
  int ws = wblk * 7 + iw + 3; if (ws >= 56) ws -= 56;
  const float* src = x + ((((size_t)b * 8 + ds) * 56 + hs) * 56 + ws) * C_DIM + lane * 4;
  float4 v = *(const float4*)src;
  float s  = v.x + v.y + v.z + v.w;
  float ss = v.x * v.x + v.y * v.y + v.z * v.z + v.w * v.w;
#pragma unroll
  for (int m = 1; m < 32; m <<= 1) { s += __shfl_xor(s, m, 32); ss += __shfl_xor(ss, m, 32); }
  float mu = s * (1.0f / 128.0f);
  float rstd = rsqrtf(ss * (1.0f / 128.0f) - mu * mu + 1e-5f);
  float4 gg = *(const float4*)(g + lane * 4);
  float4 bb = *(const float4*)(bt + lane * 4);
  v4h o;
  o[0] = (_Float16)((v.x - mu) * rstd * gg.x + bb.x);
  o[1] = (_Float16)((v.y - mu) * rstd * gg.y + bb.y);
  o[2] = (_Float16)((v.z - mu) * rstd * gg.z + bb.z);
  o[3] = (_Float16)((v.w - mu) * rstd * gg.w + bb.w);
  *(v4h*)dst = o;
}

// --------- kernel 2: QKV GEMM, scatter q (scaled) / k / v-transposed --------
// Block: 8 waves, each wave = 32 rows x 16 cols; grid.x over 32-row tiles.
__global__ void k_qkv(const _Float16* __restrict__ xw,
                      const _Float16* __restrict__ wqT,
                      const float* __restrict__ qkv_b,
                      _Float16* __restrict__ q, _Float16* __restrict__ k,
                      _Float16* __restrict__ vT) {
  int lane = threadIdx.x & 31, wv = threadIdx.x >> 5;
  int mt = blockIdx.x;                          // 0..3583 (NWIN*NPAD/32)
  int n0 = blockIdx.y * 128 + wv * 16;          // output column tile
  const _Float16* A0 = xw + (size_t)mt * 32 * C_DIM;
  const _Float16* B  = wqT + (size_t)n0 * C_DIM;
  v8f acc[2] = {};
#pragma unroll
  for (int kk = 0; kk < C_DIM; kk += 32) {
    v16h bf = ld_frag(B + kk, C_DIM, lane);
    acc[0] = wmma16(ld_frag(A0 + kk, C_DIM, lane), bf, acc[0]);
    acc[1] = wmma16(ld_frag(A0 + 16 * C_DIM + kk, C_DIM, lane), bf, acc[1]);
  }
  int col = n0 + (lane & 15);
  int part = col >> 7, hh = (col >> 5) & 3, d = col & 31;
  float bias = qkv_b[col];
  int rl0 = (lane >> 4) << 3;
#pragma unroll
  for (int sub = 0; sub < 2; ++sub) {
    int rt = mt * 2 + sub;                      // 16-row tile index
    int win = rt / 7;
    int m0 = (rt - win * 7) * 16 + rl0;
    if (part == 0) {
      size_t base = (((size_t)win * NHEADS + hh) * NPAD) * HD + d;
#pragma unroll
      for (int r = 0; r < 8; ++r)
        q[base + (size_t)(m0 + r) * HD] =
            (_Float16)((acc[sub][r] + bias) * 0.17677669529663687f);
    } else if (part == 1) {
      size_t base = (((size_t)win * NHEADS + hh) * NPAD) * HD + d;
#pragma unroll
      for (int r = 0; r < 8; ++r)
        k[base + (size_t)(m0 + r) * HD] = (_Float16)(acc[sub][r] + bias);
    } else {
      size_t base = (((size_t)win * NHEADS + hh) * HD + d) * NPAD + m0;
#pragma unroll
      for (int r = 0; r < 8; ++r)
        vT[base + r] = (_Float16)(acc[sub][r] + bias);
    }
  }
}

// -------- kernel 3: attention per (window, head), softmax + two WMMAs -------
__global__ void k_attn(const _Float16* __restrict__ q,
                       const _Float16* __restrict__ k,
                       const _Float16* __restrict__ vT,
                       const int* __restrict__ rel_idx,
                       const float* __restrict__ rpb,
                       _Float16* __restrict__ ao) {
  __shared__ __align__(16) _Float16 lds[7][16][128];
  int lane = threadIdx.x & 31, wv = threadIdx.x >> 5;
  if (wv >= 7) return;                        // 7 row-tiles of 16; no barriers used
  int win = blockIdx.x >> 2, hh = blockIdx.x & 3;
  const _Float16* qb = q + (((size_t)win * NHEADS + hh) * NPAD + wv * 16) * HD;
  const _Float16* kb = k + (((size_t)win * NHEADS + hh) * NPAD) * HD;

  v16h afrag = ld_frag(qb, HD, lane);         // K = hd = 32: one WMMA per tile
  v8f S[7];
#pragma unroll
  for (int t = 0; t < 7; ++t) {
    v8f z = {};
    S[t] = wmma16(afrag, ld_frag(kb + (size_t)t * 16 * HD, HD, lane), z);
  }
  // relative position bias + key mask
  int rl0 = (lane >> 4) << 3;
  int nb = wv * 16 + rl0, mc = lane & 15;
#pragma unroll
  for (int t = 0; t < 7; ++t) {
    int m = t * 16 + mc;
#pragma unroll
    for (int r = 0; r < 8; ++r) {
      float val = S[t][r];
      if (m < NTOK) {
        int n = nb + r;
        if (n < NTOK) val += rpb[rel_idx[n * NTOK + m] * NHEADS + hh];
      } else val = -1e30f;
      S[t][r] = val;
    }
  }
  // row softmax: reduce across the 7 register tiles, then across 16-lane group
  float mx[8], sm[8];
#pragma unroll
  for (int r = 0; r < 8; ++r) {
    float m = S[0][r];
#pragma unroll
    for (int t = 1; t < 7; ++t) m = fmaxf(m, S[t][r]);
    mx[r] = m;
  }
#pragma unroll
  for (int msk = 1; msk < 16; msk <<= 1)
#pragma unroll
    for (int r = 0; r < 8; ++r) mx[r] = fmaxf(mx[r], __shfl_xor(mx[r], msk, 32));
#pragma unroll
  for (int t = 0; t < 7; ++t)
#pragma unroll
    for (int r = 0; r < 8; ++r) S[t][r] = __expf(S[t][r] - mx[r]);
#pragma unroll
  for (int r = 0; r < 8; ++r) {
    float s = 0.f;
#pragma unroll
    for (int t = 0; t < 7; ++t) s += S[t][r];
    sm[r] = s;
  }
#pragma unroll
  for (int msk = 1; msk < 16; msk <<= 1)
#pragma unroll
    for (int r = 0; r < 8; ++r) sm[r] += __shfl_xor(sm[r], msk, 32);
#pragma unroll
  for (int r = 0; r < 8; ++r) sm[r] = 1.0f / sm[r];

  // stage P (f16) into this wave's LDS slab for A-fragment reloads
  _Float16* pl = &lds[wv][0][0];
#pragma unroll
  for (int t = 0; t < 7; ++t)
#pragma unroll
    for (int r = 0; r < 8; ++r)
      pl[(rl0 + r) * 128 + t * 16 + mc] = (_Float16)(S[t][r] * sm[r]);
#pragma unroll
  for (int r = 0; r < 8; ++r)               // zero padded K columns 112..127
    pl[(rl0 + r) * 128 + 112 + mc] = (_Float16)0.0f;

  // O = P @ V ; V stored transposed [d][m] so B frags are contiguous
  const _Float16* vb = vT + (((size_t)win * NHEADS + hh) * HD) * NPAD;
#pragma unroll
  for (int dt = 0; dt < 2; ++dt) {
    v8f O = {};
#pragma unroll
    for (int kc = 0; kc < 4; ++kc)
      O = wmma16(ld_frag(pl + kc * 32, 128, lane),
                 ld_frag(vb + (size_t)dt * 16 * NPAD + kc * 32, NPAD, lane), O);
    size_t rb = ((size_t)win * NPAD + wv * 16 + rl0) * C_DIM + hh * HD + dt * 16 + mc;
#pragma unroll
    for (int r = 0; r < 8; ++r) ao[rb + (size_t)r * C_DIM] = (_Float16)O[r];
  }
}

// ------ kernel 4: proj GEMM + window reverse + roll(+1,+3,+3) + residual ----
__global__ void k_proj(const _Float16* __restrict__ ao,
                       const _Float16* __restrict__ wpT,
                       const float* __restrict__ proj_b,
                       const float* __restrict__ x, float* __restrict__ y) {
  int lane = threadIdx.x & 31, wv = threadIdx.x >> 5;
  int mt = blockIdx.x;                          // 0..3583
  const _Float16* A0 = ao + (size_t)mt * 32 * C_DIM;
  const _Float16* B  = wpT + (size_t)(wv * 16) * C_DIM;
  v8f acc[2] = {};
#pragma unroll
  for (int kk = 0; kk < C_DIM; kk += 32) {
    v16h bf = ld_frag(B + kk, C_DIM, lane);
    acc[0] = wmma16(ld_frag(A0 + kk, C_DIM, lane), bf, acc[0]);
    acc[1] = wmma16(ld_frag(A0 + 16 * C_DIM + kk, C_DIM, lane), bf, acc[1]);
  }
  int col = wv * 16 + (lane & 15);
  float pb = proj_b[col];
  int rl0 = (lane >> 4) << 3;
#pragma unroll
  for (int sub = 0; sub < 2; ++sub) {
    int rt = mt * 2 + sub;
    int win = rt / 7;
    int p0 = (rt - win * 7) * 16 + rl0;
    int b = win >> 8, rem = win & 255;
    int dblk = rem >> 6, hblk = (rem >> 3) & 7, wblk = rem & 7;
#pragma unroll
    for (int r = 0; r < 8; ++r) {
      int p = p0 + r;
      if (p >= NTOK) continue;               // drop padded rows
      int id = p / 49, r2 = p - id * 49, ih = r2 / 7, iw = r2 - ih * 7;
      int dd = (dblk * 2 + id + 1) & 7;
      int hh = hblk * 7 + ih + 3; if (hh >= 56) hh -= 56;
      int ww = wblk * 7 + iw + 3; if (ww >= 56) ww -= 56;
      size_t idx = ((((size_t)b * 8 + dd) * 56 + hh) * 56 + ww) * C_DIM + col;
      y[idx] = x[idx] + acc[sub][r] + pb;
    }
  }
}

// -------------------------- kernel 5: LN2 -> f16 ----------------------------
__global__ void k_ln2(const float* __restrict__ y, const float* __restrict__ g,
                      const float* __restrict__ bt, _Float16* __restrict__ out) {
  int gid = blockIdx.x * 8 + (threadIdx.x >> 5);
  int lane = threadIdx.x & 31;
  const float* src = y + (size_t)gid * C_DIM + lane * 4;
  float4 v = *(const float4*)src;
  float s = v.x + v.y + v.z + v.w;
  float ss = v.x * v.x + v.y * v.y + v.z * v.z + v.w * v.w;
#pragma unroll
  for (int m = 1; m < 32; m <<= 1) { s += __shfl_xor(s, m, 32); ss += __shfl_xor(ss, m, 32); }
  float mu = s * (1.0f / 128.0f);
  float rstd = rsqrtf(ss * (1.0f / 128.0f) - mu * mu + 1e-5f);
  float4 gg = *(const float4*)(g + lane * 4);
  float4 bb = *(const float4*)(bt + lane * 4);
  v4h o;
  o[0] = (_Float16)((v.x - mu) * rstd * gg.x + bb.x);
  o[1] = (_Float16)((v.y - mu) * rstd * gg.y + bb.y);
  o[2] = (_Float16)((v.z - mu) * rstd * gg.z + bb.z);
  o[3] = (_Float16)((v.w - mu) * rstd * gg.w + bb.w);
  *(v4h*)(out + (size_t)gid * C_DIM + lane * 4) = o;
}

// ------------------- kernel 6: MLP1 GEMM + exact GELU -----------------------
__global__ void k_mlp1(const _Float16* __restrict__ h, const _Float16* __restrict__ w1T,
                       const float* __restrict__ b1, _Float16* __restrict__ h1) {
  int lane = threadIdx.x & 31, wv = threadIdx.x >> 5;
  int mt = blockIdx.x;                          // 0..3135 (TT/32)
  int n0 = blockIdx.y * 128 + wv * 16;
  const _Float16* A0 = h + (size_t)mt * 32 * C_DIM;
  const _Float16* B  = w1T + (size_t)n0 * C_DIM;
  v8f acc[2] = {};
#pragma unroll
  for (int kk = 0; kk < C_DIM; kk += 32) {
    v16h bf = ld_frag(B + kk, C_DIM, lane);
    acc[0] = wmma16(ld_frag(A0 + kk, C_DIM, lane), bf, acc[0]);
    acc[1] = wmma16(ld_frag(A0 + 16 * C_DIM + kk, C_DIM, lane), bf, acc[1]);
  }
  int col = n0 + (lane & 15);
  float bias = b1[col];
  int rl0 = (lane >> 4) << 3;
#pragma unroll
  for (int sub = 0; sub < 2; ++sub)
#pragma unroll
    for (int r = 0; r < 8; ++r) {
      float t = acc[sub][r] + bias;
      float ge = 0.5f * t * (1.0f + erff(t * 0.70710678118f));
      h1[(size_t)(mt * 32 + sub * 16 + rl0 + r) * MLPH + col] = (_Float16)ge;
    }
}

// ---------------------- kernel 7: MLP2 GEMM + residual ----------------------
__global__ void k_mlp2(const _Float16* __restrict__ h1, const _Float16* __restrict__ w2T,
                       const float* __restrict__ b2, const float* __restrict__ y,
                       float* __restrict__ out) {
  int lane = threadIdx.x & 31, wv = threadIdx.x >> 5;
  int mt = blockIdx.x;                          // 0..3135
  const _Float16* A0 = h1 + (size_t)mt * 32 * MLPH;
  const _Float16* B  = w2T + (size_t)(wv * 16) * MLPH;
  v8f acc[2] = {};
#pragma unroll
  for (int kk = 0; kk < MLPH; kk += 32) {
    __builtin_prefetch(A0 + kk + 128, 0, 0);  // global_prefetch_b8 ahead in K
    v16h bf = ld_frag(B + kk, MLPH, lane);
    acc[0] = wmma16(ld_frag(A0 + kk, MLPH, lane), bf, acc[0]);
    acc[1] = wmma16(ld_frag(A0 + 16 * MLPH + kk, MLPH, lane), bf, acc[1]);
  }
  int col = wv * 16 + (lane & 15);
  float bias = b2[col];
  int rl0 = (lane >> 4) << 3;
#pragma unroll
  for (int sub = 0; sub < 2; ++sub)
#pragma unroll
    for (int r = 0; r < 8; ++r) {
      size_t idx = (size_t)(mt * 32 + sub * 16 + rl0 + r) * C_DIM + col;
      out[idx] = y[idx] + acc[sub][r] + bias;
    }
}

// ---------------------------------------------------------------------------
extern "C" void kernel_launch(void* const* d_in, const int* in_sizes, int n_in,
                              void* d_out, int out_size, void* d_ws, size_t ws_size,
                              hipStream_t stream) {
  const float* x      = (const float*)d_in[0];
  const int*   relidx = (const int*)  d_in[1];
  const float* rpb    = (const float*)d_in[2];
  const float* qkv_w  = (const float*)d_in[3];
  const float* qkv_b  = (const float*)d_in[4];
  const float* proj_w = (const float*)d_in[5];
  const float* proj_b = (const float*)d_in[6];
  const float* ln1_g  = (const float*)d_in[7];
  const float* ln1_b  = (const float*)d_in[8];
  const float* ln2_g  = (const float*)d_in[9];
  const float* ln2_b  = (const float*)d_in[10];
  const float* w1     = (const float*)d_in[11];
  const float* b1     = (const float*)d_in[12];
  const float* w2     = (const float*)d_in[13];
  const float* b2     = (const float*)d_in[14];

  char* ws = (char*)d_ws;
  const size_t SZ_WQ  = (size_t)384 * 128 * 2;
  const size_t SZ_WP  = (size_t)128 * 128 * 2;
  const size_t SZ_W1  = (size_t)512 * 128 * 2;
  const size_t SZ_W2  = (size_t)128 * 512 * 2;
  const size_t SZ_BIG = (size_t)NWIN * NPAD * C_DIM * 2;   // 29,360,128 B
  size_t off = 0;
  auto take = [&](size_t n) { size_t r = off; off += (n + 255) & ~(size_t)255; return r; };
  size_t o_wq = take(SZ_WQ), o_wp = take(SZ_WP), o_w1 = take(SZ_W1), o_w2 = take(SZ_W2);
  size_t o_xw = take(SZ_BIG);     // xw, reused as ao, start of h1 overlay
  size_t o_q  = take(SZ_BIG);
  size_t o_k  = take(SZ_BIG);
  size_t o_vT = take(SZ_BIG);
  size_t o_y   = take((size_t)TT * C_DIM * 4);
  size_t o_ln2 = take((size_t)TT * C_DIM * 2);
  size_t o_ao = o_xw;             // dead xw after QKV
  size_t o_h1 = o_xw;             // h1 (102.8MB) overlays xw..vT (117.4MB), dead after attn/proj

  _Float16* wqT = (_Float16*)(ws + o_wq);
  _Float16* wpT = (_Float16*)(ws + o_wp);
  _Float16* w1T = (_Float16*)(ws + o_w1);
  _Float16* w2T = (_Float16*)(ws + o_w2);
  _Float16* xw  = (_Float16*)(ws + o_xw);
  _Float16* qh  = (_Float16*)(ws + o_q);
  _Float16* kh  = (_Float16*)(ws + o_k);
  _Float16* vT  = (_Float16*)(ws + o_vT);
  _Float16* ao  = (_Float16*)(ws + o_ao);
  float*    y   = (float*)   (ws + o_y);
  _Float16* l2h = (_Float16*)(ws + o_ln2);
  _Float16* h1  = (_Float16*)(ws + o_h1);

  // 0) weights -> transposed f16
  k_transpose_w<<<(384 * 128 + 255) / 256, 256, 0, stream>>>(qkv_w, wqT, 128, 384);
  k_transpose_w<<<(128 * 128 + 255) / 256, 256, 0, stream>>>(proj_w, wpT, 128, 128);
  k_transpose_w<<<(512 * 128 + 255) / 256, 256, 0, stream>>>(w1, w1T, 128, 512);
  k_transpose_w<<<(128 * 512 + 255) / 256, 256, 0, stream>>>(w2, w2T, 512, 128);
  // 1) LN1 + shift + window partition (8 waves/block, 1 token each)
  k_ln1_window<<<NWIN * NPAD / 8, 256, 0, stream>>>(x, ln1_g, ln1_b, xw);
  // 2) QKV GEMM: M=NWIN*NPAD, K=128, N=384 (32-row tiles per block)
  k_qkv<<<dim3(NWIN * NPAD / 32, 3), 256, 0, stream>>>(xw, wqT, qkv_b, qh, kh, vT);
  // 3) attention per (window, head)
  k_attn<<<NWIN * NHEADS, 256, 0, stream>>>(qh, kh, vT, relidx, rpb, ao);
  // 4) proj GEMM + reverse + residual
  k_proj<<<NWIN * NPAD / 32, 256, 0, stream>>>(ao, wpT, proj_b, x, y);
  // 5) LN2
  k_ln2<<<TT / 8, 256, 0, stream>>>(y, ln2_g, ln2_b, l2h);
  // 6) MLP1 + GELU: M=TT, K=128, N=512
  k_mlp1<<<dim3(TT / 32, 4), 256, 0, stream>>>(l2h, w1T, b1, h1);
  // 7) MLP2 + residual: M=TT, K=512, N=128
  k_mlp2<<<TT / 32, 256, 0, stream>>>(h1, w2T, b2, y, (float*)d_out);
  (void)in_sizes; (void)n_in; (void)out_size; (void)ws_size;
}
